// LinearProjection_11089605558541
// MI455X (gfx1250) — matrix operand order repaired and checked by reading, not compiled, not run on hardware
//
#include <hip/hip_runtime.h>
#include <stdint.h>

typedef __bf16   v16bf __attribute__((ext_vector_type(16)));
typedef float    v8f   __attribute__((ext_vector_type(8)));
typedef uint32_t v8u   __attribute__((ext_vector_type(8)));
typedef uint32_t u32x4 __attribute__((ext_vector_type(4)));
typedef int32_t  i32x8 __attribute__((ext_vector_type(8)));
typedef int32_t  i32x4 __attribute__((ext_vector_type(4)));

#define B_    64
#define N_    2048
#define M_TOT (B_ * N_)        // 131072 rows
#define EMB   512
#define VIS   18
#define BBOX  4
#define KPTC  51               // 17*3
#define FEAT  585              // EMB+VIS+BBOX+KPTC
#define NKT   19               // k-tiles of 32 (K padded to 608)
#define TOK   1024
#define WP_DW (NKT * TOK * 16) // packed W dwords: [kt][t][kp] = 311296

#if __has_builtin(__builtin_amdgcn_tensor_load_to_lds) && \
    __has_builtin(__builtin_amdgcn_s_wait_tensorcnt)
#define USE_TDM 1
#else
#define USE_TDM 0
#endif

__device__ __forceinline__ uint32_t pack_bf16x2(float a, float b) {
  __bf16 x = (__bf16)a, y = (__bf16)b;          // native v_cvt, RNE
  return (uint32_t)__builtin_bit_cast(uint16_t, x) |
         ((uint32_t)__builtin_bit_cast(uint16_t, y) << 16);
}

// Repack W [TOK x FEAT] f32 -> bf16 in WMMA-B-fragment-major order:
// Wp dword index  kt*16384 + t*16 + kp  holds { bf16(W[t][kt*32+2kp]),
// bf16(W[t][kt*32+2kp+1]) }, zero-padded for k >= FEAT.  Per-block B tile is
// one contiguous 8KB chunk; each lane's fragment is two 16B LDS runs.
__global__ void pack_w_bf16(const float* __restrict__ W, uint32_t* __restrict__ Wp) {
  int idx = blockIdx.x * blockDim.x + threadIdx.x;
  if (idx >= WP_DW) return;
  int kp = idx & 15;
  int t  = (idx >> 4) & 1023;
  int kt = idx >> 14;
  int k0 = kt * 32 + 2 * kp;
  float f0 = (k0     < FEAT) ? W[t * FEAT + k0]     : 0.0f;
  float f1 = (k0 + 1 < FEAT) ? W[t * FEAT + k0 + 1] : 0.0f;
  Wp[idx] = pack_bf16x2(f0, f1);
}

#if USE_TDM
// Issue a 1-D TDM copy: 8KB (1024 x 8B elements) global -> LDS.
// D# per CDNA5 ISA §8.3/8.4; descriptor is wave-uniform -> lives in SGPRs.
// clang-23 toolchain arity: (u32x4 g0, i32x8 g1, i32x4, i32x4, i32x8, i32 cpol)
__device__ __forceinline__ void tdm_load_8k(const uint32_t* gsrc, uint32_t* ldst) {
  uint64_t ga = (uint64_t)(uintptr_t)gsrc;
  uint32_t la = (uint32_t)(uintptr_t)ldst;      // low 32 bits = LDS byte addr
  u32x4 g0;
  g0[0] = 1u;                                   // count=1, user mode, no gather
  g0[1] = la;                                   // lds_addr
  g0[2] = (uint32_t)ga;                         // global_addr[31:0]
  g0[3] = (uint32_t)((ga >> 32) & 0x01FFFFFFu)  // global_addr[56:32]
          | (2u << 30);                         // type = 2 ("image")
  i32x8 g1;
  g1[0] = (int32_t)(3u << 16);                  // data_size=8B; mask/flags=0
  g1[1] = (int32_t)(1024u << 16);               // tensor_dim0[15:0]=1024
  g1[2] = (int32_t)(1u << 16);                  // tensor_dim0 hi=0; tensor_dim1=1
  g1[3] = (int32_t)(1024u << 16);               // tensor_dim1 hi=0; tile_dim0=1024
  g1[4] = 0;                                    // tile_dim1=0, tile_dim2=0 (1-D)
  g1[5] = 1024;                                 // tensor_dim0_stride lo (unused)
  g1[6] = 0;
  g1[7] = 0;
  i32x4 gz4 = {0, 0, 0, 0};                     // groups 2/3: tensor <= 2D
  i32x8 gz8 = {0, 0, 0, 0, 0, 0, 0, 0};
  __builtin_amdgcn_tensor_load_to_lds(g0, g1, gz4, gz4, gz8, 0);
}
#endif

// Fused concat + GEMM (bf16 WMMA, f32 accum) + bias + row-mask.
// Block: 256 thr (8 waves), tile 128(M) x 128(T); wave = 64x32 = 4x2 accs.
// Double-buffered LDS pipeline; B tile moved by the Tensor Data Mover.
__global__ __launch_bounds__(256)
void fused_concat_linear_wmma(const float* __restrict__ emb,
                              const float* __restrict__ vis,
                              const float* __restrict__ bbox,
                              const float* __restrict__ kpt,
                              const unsigned char* __restrict__ mask,
                              const uint32_t* __restrict__ Wp,
                              const float* __restrict__ bias,
                              float* __restrict__ out) {
  __shared__ __align__(16) uint16_t sA[2][128 * 32];  // [buf][row][k]   bf16
  __shared__ __align__(16) uint32_t sB[2][128 * 16];  // [buf][t][kp]    bf16x2

  const int tid      = threadIdx.x;
  const int lane     = tid & 31;
  const int wid      = tid >> 5;             // 0..7
  const int waveM    = (wid >> 2) * 64;      // 0 or 64
  const int waveT    = (wid & 3) * 32;       // 0,32,64,96
  const int laneHalf = lane >> 4;            // 0 or 1
  const int lm       = lane & 15;

  const int M0 = blockIdx.x * 128;
  const int T0 = blockIdx.y * 128;

  auto loadTile = [&](int kt, int buf) {
    if (kt < 16) {
      // -- pure-embeddings fast path: coalesced float4 stream, cvt, b64 store
#pragma unroll
      for (int i = 0; i < 4; ++i) {
        int g    = tid + i * 256;            // 1024 float4 groups per tile
        int r    = g >> 3;
        int koff = (g & 7) * 4;
        const float4 f =
            *(const float4*)(emb + (size_t)(M0 + r) * EMB + kt * 32 + koff);
        uint2 p;
        p.x = pack_bf16x2(f.x, f.y);
        p.y = pack_bf16x2(f.z, f.w);
        *(uint2*)&sA[buf][r * 32 + koff] = p;
      }
      if (kt + 1 < 16)                        // prefetch next emb tile
        __builtin_prefetch(emb + (size_t)(M0 + (tid >> 3)) * EMB +
                               (kt + 1) * 32 + (tid & 7) * 4, 0, 3);
    } else {
      // -- tail gather: vis/bbox/kpt + zero pad (3 of 19 tiles)
#pragma unroll 4
      for (int i = 0; i < 16; ++i) {
        int idx = tid + i * 256;
        int r   = idx >> 5;
        int k   = (idx & 31) + kt * 32;      // 512..607
        int m   = M0 + r;
        float f;
        if (k < EMB + VIS)              f = vis[(size_t)m * VIS + (k - EMB)];
        else if (k < EMB + VIS + BBOX)  f = bbox[(size_t)m * BBOX + (k - EMB - VIS)];
        else if (k < FEAT)              f = kpt[(size_t)m * KPTC + (k - EMB - VIS - BBOX)];
        else                            f = 0.0f;
        sA[buf][idx] = (uint16_t)(pack_bf16x2(f, 0.0f) & 0xFFFFu);
      }
    }
#if USE_TDM
    // -- B tile: one 8KB DMA via the Tensor Data Mover (wave 0 only; the
    //    descriptor is uniform, TENSORcnt drained before the barrier).
    if (wid == 0)
      tdm_load_8k(Wp + (size_t)kt * 16384 + (size_t)T0 * 16, &sB[buf][0]);
#else
    // -- fallback: cooperative contiguous uint4 copy
#pragma unroll
    for (int i = 0; i < 2; ++i) {
      int jj = tid + i * 256;                // 512 uint4s
      uint4 v = ((const uint4*)Wp)[(size_t)kt * 4096 + T0 * 4 + jj];
      ((uint4*)sB[buf])[jj] = v;
    }
#endif
  };

  v8f acc[4][2];
  const v8f vzero = {0.f, 0.f, 0.f, 0.f, 0.f, 0.f, 0.f, 0.f};
#pragma unroll
  for (int i = 0; i < 4; ++i)
#pragma unroll
    for (int j = 0; j < 2; ++j) acc[i][j] = vzero;

  loadTile(0, 0);
#if USE_TDM
  if (wid == 0) __builtin_amdgcn_s_wait_tensorcnt(0);
#endif
  __syncthreads();

  for (int kt = 0; kt < NKT; ++kt) {
    const int cur = kt & 1;
    if (kt + 1 < NKT) loadTile(kt + 1, cur ^ 1);

    // ---- fragments per ISA VGPR layouts ----
    v16bf afrag[4];
#pragma unroll
    for (int mi = 0; mi < 4; ++mi) {
      int row = waveM + mi * 16 + lm;        // lane = M in 16x32 A tile
      v8u au;
#pragma unroll
      for (int v = 0; v < 8; ++v) {
        int k0 = (v < 4 ? 2 * v : 16 + 2 * (v - 4)) + laneHalf * 8;
        au[v] = *(const uint32_t*)&sA[cur][row * 32 + k0];
      }
      afrag[mi] = __builtin_bit_cast(v16bf, au);
    }
    v16bf bfrag[2];
#pragma unroll
    for (int ni = 0; ni < 2; ++ni) {
      int t = waveT + ni * 16 + lm;          // lane = N; two 16B runs per frag
      v8u bu;
#pragma unroll
      for (int v = 0; v < 8; ++v)
        bu[v] = sB[cur][t * 16 + laneHalf * 4 + (v < 4 ? v : v + 4)];
      bfrag[ni] = __builtin_bit_cast(v16bf, bu);
    }
#pragma unroll
    for (int mi = 0; mi < 4; ++mi)
#pragma unroll
      for (int ni = 0; ni < 2; ++ni)
        acc[mi][ni] = __builtin_amdgcn_wmma_f32_16x16x32_bf16(
            false, afrag[mi], false, bfrag[ni], (short)0, acc[mi][ni],
            false, false);
#if USE_TDM
    if (wid == 0) __builtin_amdgcn_s_wait_tensorcnt(0);  // drain DMA for kt+1
#endif
    __syncthreads();
  }

  // ---- epilogue: + bias, row mask, f32 stores ----
  const float bt[2] = {bias[T0 + waveT + lm], bias[T0 + waveT + 16 + lm]};
#pragma unroll
  for (int mi = 0; mi < 4; ++mi) {
    int rowBase = M0 + waveM + mi * 16 + laneHalf * 8;   // C/D layout
    uint2 mraw  = *(const uint2*)(mask + rowBase);       // 8 mask bytes
    uint64_t mb = ((uint64_t)mraw.y << 32) | (uint64_t)mraw.x;
#pragma unroll
    for (int ni = 0; ni < 2; ++ni) {
      int col = T0 + waveT + ni * 16 + lm;
#pragma unroll
      for (int r = 0; r < 8; ++r) {
        float v = acc[mi][ni][r] + bt[ni];
        out[(size_t)(rowBase + r) * TOK + col] =
            ((mb >> (8 * r)) & 1ull) ? v : 0.0f;
      }
    }
  }
}

extern "C" void kernel_launch(void* const* d_in, const int* in_sizes, int n_in,
                              void* d_out, int out_size, void* d_ws, size_t ws_size,
                              hipStream_t stream) {
  const float*         emb  = (const float*)d_in[0];
  const float*         vis  = (const float*)d_in[1];
  const float*         bbox = (const float*)d_in[2];
  const float*         kpt  = (const float*)d_in[3];
  const unsigned char* mask = (const unsigned char*)d_in[4];  // jnp.bool_
  const float*         W    = (const float*)d_in[5];
  const float*         bvec = (const float*)d_in[6];
  float*               out  = (float*)d_out;
  uint32_t*            Wp   = (uint32_t*)d_ws;  // 1.22 MB scratch

  pack_w_bf16<<<(WP_DW + 255) / 256, 256, 0, stream>>>(W, Wp);

  dim3 grid(M_TOT / 128, TOK / 128);            // 1024 x 8 blocks
  fused_concat_linear_wmma<<<grid, 256, 0, stream>>>(emb, vis, bbox, kpt, mask,
                                                     Wp, bvec, out);
}
